// BiMambaBlock_62989990363685
// MI455X (gfx1250) — matrix operand order repaired and checked by reading, not compiled, not run on hardware
//
#include <hip/hip_runtime.h>
#include <cstdint>

// ---------------- problem constants ----------------
#define BB       2
#define LL       2048
#define MROWS    (BB * LL)          // 4096
#define DMODEL   1024
#define DHALF    512
#define DINNER   1024
#define DSTATE   128
#define DCONV    4
#define NHEADS   16
#define HEADDIM  64
#define CONVDIM  (DINNER + 2 * DSTATE)               // 1280
#define DINPROJ  (2 * DINNER + 2 * DSTATE + NHEADS)  // 2320
#define QCHUNK   64
#define NCHUNK   (LL / QCHUNK)                       // 32
#define NBATCH   (BB * NCHUNK * NHEADS)              // 1024
#define EPSF     1e-5f

// ---------------- vector types ----------------
typedef __attribute__((ext_vector_type(16))) __bf16 v16bf;
typedef __attribute__((ext_vector_type(8)))  __bf16 v8bf;
typedef __attribute__((ext_vector_type(8)))  float  v8f;

static __device__ __forceinline__ __bf16 tobf(float f) {
  unsigned u = __builtin_bit_cast(unsigned, f);
  unsigned r = (u + 0x7FFFu + ((u >> 16) & 1u)) >> 16;   // RNE
  unsigned short s = (unsigned short)r;
  return __builtin_bit_cast(__bf16, s);
}

static __device__ __forceinline__ float block_reduce_sum(float v, float* red) {
  int tid = threadIdx.x;
  red[tid] = v;
  __syncthreads();
  for (int s = 128; s > 0; s >>= 1) {
    if (tid < s) red[tid] += red[tid + s];
    __syncthreads();
  }
  float r = red[0];
  __syncthreads();
  return r;
}

static __device__ __forceinline__ v8f wmma_bf16(v16bf a, v16bf b, v8f c) {
  return __builtin_amdgcn_wmma_f32_16x16x32_bf16(
      false, a, false, b, (short)0, c, false, false);
}

// ---------------- K1: rmsnorm over D_MODEL, split halves, time-flip backward half
__global__ __launch_bounds__(256)
void rmsnorm_split(const float* __restrict__ x, const float* __restrict__ w,
                   __bf16* __restrict__ uf, __bf16* __restrict__ ub) {
  __shared__ float red[256];
  const int m = blockIdx.x, tid = threadIdx.x;
  const int b = m / LL, l = m % LL;
  float ss = 0.f;
  for (int i = tid; i < DMODEL; i += 256) {
    float v = x[(size_t)m * DMODEL + i];
    ss += v * v;
  }
  float tot = block_reduce_sum(ss, red);
  float sc = rsqrtf(tot / (float)DMODEL + EPSF);
  const size_t rf = (size_t)m * DHALF;
  const size_t rb = ((size_t)b * LL + (LL - 1 - l)) * (size_t)DHALF;
  for (int i = tid; i < DHALF; i += 256) {
    uf[rf + i] = tobf(x[(size_t)m * DMODEL + i] * sc * w[i]);
    ub[rb + i] = tobf(x[(size_t)m * DMODEL + DHALF + i] * sc * w[DHALF + i]);
  }
}

// ---------------- K2: f32 -> bf16 convert
__global__ __launch_bounds__(256)
void cvt_bf16(const float* __restrict__ in, __bf16* __restrict__ out, size_t n) {
  for (size_t i = blockIdx.x * (size_t)blockDim.x + threadIdx.x; i < n;
       i += (size_t)gridDim.x * blockDim.x)
    out[i] = tobf(in[i]);
}

// ---------------- K3: WMMA GEMM  C[M,N](f32) = A[M,K](bf16) * W[N,K](bf16)^T
__global__ __launch_bounds__(128)
void gemm_bf16(const __bf16* __restrict__ A, int lda,
               const __bf16* __restrict__ Bw, int ldb,
               float* __restrict__ C, int ldc,
               int N, int K,
               const float* __restrict__ bias,
               const float* __restrict__ resid,
               int col_off, int flip) {
  const int lane = threadIdx.x & 31;
  const int w    = threadIdx.x >> 5;
  const int n0   = lane & 15;
  const int hi   = lane >> 4;
  const int Mtile = blockIdx.y * 64 + (w >> 1) * 32;
  const int Ntile = blockIdx.x * 64 + (w & 1) * 32;

  v8f acc00 = {}, acc01 = {}, acc10 = {}, acc11 = {};

  const __bf16* aBase0 = A + (size_t)(Mtile + n0) * lda;
  const __bf16* aBase1 = A + (size_t)(Mtile + 16 + n0) * lda;
  int nc0 = Ntile + n0;       if (nc0 >= N) nc0 = N - 1;
  int nc1 = Ntile + 16 + n0;  if (nc1 >= N) nc1 = N - 1;
  const __bf16* bBase0 = Bw + (size_t)nc0 * ldb + hi * 16;
  const __bf16* bBase1 = Bw + (size_t)nc1 * ldb + hi * 16;

  for (int k = 0; k < K; k += 32) {
    v8bf a0lo = *(const v8bf*)(aBase0 + k + hi * 16);
    v8bf a0hi = *(const v8bf*)(aBase0 + k + 8 + hi * 16);
    v8bf a1lo = *(const v8bf*)(aBase1 + k + hi * 16);
    v8bf a1hi = *(const v8bf*)(aBase1 + k + 8 + hi * 16);
    v16bf a0 = __builtin_shufflevector(a0lo, a0hi, 0,1,2,3,4,5,6,7,8,9,10,11,12,13,14,15);
    v16bf a1 = __builtin_shufflevector(a1lo, a1hi, 0,1,2,3,4,5,6,7,8,9,10,11,12,13,14,15);
    v16bf b0 = *(const v16bf*)(bBase0 + k);
    v16bf b1 = *(const v16bf*)(bBase1 + k);
    acc00 = wmma_bf16(a0, b0, acc00);
    acc01 = wmma_bf16(a0, b1, acc01);
    acc10 = wmma_bf16(a1, b0, acc10);
    acc11 = wmma_bf16(a1, b1, acc11);
  }

  #pragma unroll
  for (int sm = 0; sm < 2; ++sm) {
    #pragma unroll
    for (int sn = 0; sn < 2; ++sn) {
      v8f acc = sm ? (sn ? acc11 : acc10) : (sn ? acc01 : acc00);
      int gn = Ntile + sn * 16 + n0;
      if (gn >= N) continue;
      #pragma unroll
      for (int r = 0; r < 8; ++r) {
        int gm = Mtile + sm * 16 + hi * 8 + r;
        int orow = gm;
        if (flip) {
          int bb = gm / LL, ll = gm % LL;
          orow = bb * LL + (LL - 1 - ll);
        }
        float v = acc[r];
        if (bias)  v += bias[gn];
        if (resid) v += resid[(size_t)orow * ldc + col_off + gn];
        C[(size_t)orow * ldc + col_off + gn] = v;
      }
    }
  }
}

// ---------------- K3b: batched WMMA GEMM for chunked SSD
// Per batch: 64 rows. A base = A + (batch/aDiv)*aStride, similarly B.
// C base = C + (batch/cDiv)*cStride + (batch%cMod)*cColUnit.
// Optional per-row scale exp(rowscale[batch*64+row]) and accumulate-into-C.
__global__ __launch_bounds__(128)
void gemm_bf16_batched(const __bf16* __restrict__ A, int lda, size_t aStride, int aDiv,
                       const __bf16* __restrict__ Bw, int ldb, size_t bStride, int bDiv,
                       float* __restrict__ C, int ldc, size_t cStride, int cDiv,
                       int cMod, int cColUnit,
                       int N, int K,
                       const float* __restrict__ rowscale, int accumulate) {
  const int batch = blockIdx.y;
  const int lane = threadIdx.x & 31;
  const int w    = threadIdx.x >> 5;
  const int n0   = lane & 15;
  const int hi   = lane >> 4;
  const int Mt = (w >> 1) * 32;                 // row within this 64-row batch
  const int Nt = blockIdx.x * 64 + (w & 1) * 32;

  const __bf16* Ab = A  + (size_t)(batch / aDiv) * aStride;
  const __bf16* Bb = Bw + (size_t)(batch / bDiv) * bStride;
  float* Cb = C + (size_t)(batch / cDiv) * cStride + (size_t)(batch % cMod) * cColUnit;

  v8f acc00 = {}, acc01 = {}, acc10 = {}, acc11 = {};
  const __bf16* aBase0 = Ab + (size_t)(Mt + n0) * lda;
  const __bf16* aBase1 = Ab + (size_t)(Mt + 16 + n0) * lda;
  const __bf16* bBase0 = Bb + (size_t)(Nt + n0) * ldb + hi * 16;
  const __bf16* bBase1 = Bb + (size_t)(Nt + 16 + n0) * ldb + hi * 16;

  for (int k = 0; k < K; k += 32) {
    v8bf a0lo = *(const v8bf*)(aBase0 + k + hi * 16);
    v8bf a0hi = *(const v8bf*)(aBase0 + k + 8 + hi * 16);
    v8bf a1lo = *(const v8bf*)(aBase1 + k + hi * 16);
    v8bf a1hi = *(const v8bf*)(aBase1 + k + 8 + hi * 16);
    v16bf a0 = __builtin_shufflevector(a0lo, a0hi, 0,1,2,3,4,5,6,7,8,9,10,11,12,13,14,15);
    v16bf a1 = __builtin_shufflevector(a1lo, a1hi, 0,1,2,3,4,5,6,7,8,9,10,11,12,13,14,15);
    v16bf b0 = *(const v16bf*)(bBase0 + k);
    v16bf b1 = *(const v16bf*)(bBase1 + k);
    acc00 = wmma_bf16(a0, b0, acc00);
    acc01 = wmma_bf16(a0, b1, acc01);
    acc10 = wmma_bf16(a1, b0, acc10);
    acc11 = wmma_bf16(a1, b1, acc11);
  }

  #pragma unroll
  for (int sm = 0; sm < 2; ++sm) {
    #pragma unroll
    for (int sn = 0; sn < 2; ++sn) {
      v8f acc = sm ? (sn ? acc11 : acc10) : (sn ? acc01 : acc00);
      int gn = Nt + sn * 16 + n0;
      #pragma unroll
      for (int r = 0; r < 8; ++r) {
        int rl = Mt + sm * 16 + hi * 8 + r;
        float v = acc[r];
        if (rowscale) v *= expf(rowscale[(size_t)batch * QCHUNK + rl]);
        size_t idx = (size_t)rl * ldc + gn;
        if (accumulate) v += Cb[idx];
        Cb[idx] = v;
      }
    }
  }
}

// ---------------- K4: causal depthwise conv (k=4) + silu
__global__ __launch_bounds__(256)
void conv_silu(const float* __restrict__ zx, const float* __restrict__ cw,
               const float* __restrict__ cb, float* __restrict__ xbc) {
  const int idx = blockIdx.x;          // b*L + l
  const int l = idx % LL;
  for (int c = threadIdx.x; c < CONVDIM; c += 256) {
    float acc = cb[c];
    #pragma unroll
    for (int k = 0; k < DCONV; ++k) {
      int ls = l - (DCONV - 1) + k;
      if (ls >= 0)
        acc = fmaf(zx[(size_t)(idx - (DCONV - 1) + k) * DINPROJ + DINNER + c],
                   cw[c * DCONV + k], acc);
    }
    xbc[(size_t)idx * CONVDIM + c] = acc / (1.f + expf(-acc));
  }
}

// ---------------- K5: dt = softplus(dt_raw + bias)
__global__ __launch_bounds__(256)
void dt_prep(const float* __restrict__ zx, const float* __restrict__ dt_bias,
             float* __restrict__ dt) {
  const int total = MROWS * NHEADS;
  for (int i = blockIdx.x * blockDim.x + threadIdx.x; i < total;
       i += gridDim.x * blockDim.x) {
    int m = i >> 4, h = i & 15;
    float v = zx[(size_t)m * DINPROJ + (DINNER + CONVDIM) + h] + dt_bias[h];
    dt[i] = (v > 20.f) ? v : log1pf(expf(v));
  }
}

// ---------------- K6: per-(chunk,head) inclusive cumsum of log(dA) = -exp(A_log)*dt
__global__ __launch_bounds__(256)
void chunk_cs(const float* __restrict__ dt, const float* __restrict__ A_log,
              float* __restrict__ cs) {
  int idx = blockIdx.x * blockDim.x + threadIdx.x;   // bch = chunk*NH + h
  if (idx >= NBATCH) return;
  int chunk = idx / NHEADS, h = idx % NHEADS;
  float Ac = -expf(A_log[h]);
  float acc = 0.f;
  for (int t = 0; t < QCHUNK; ++t) {
    acc += Ac * dt[((size_t)chunk * QCHUNK + t) * NHEADS + h];
    cs[(size_t)idx * QCHUNK + t] = acc;
  }
}

// ---------------- K7: extract B,C rows as bf16 [m,128]
__global__ __launch_bounds__(128)
void build_bc(const float* __restrict__ xbc, __bf16* __restrict__ Bbf,
              __bf16* __restrict__ Cbf) {
  const int m = blockIdx.x, n = threadIdx.x;
  Bbf[(size_t)m * DSTATE + n] = tobf(xbc[(size_t)m * CONVDIM + DINNER + n]);
  Cbf[(size_t)m * DSTATE + n] = tobf(xbc[(size_t)m * CONVDIM + DINNER + DSTATE + n]);
}

// ---------------- K8: per-chunk B transposed: Bt[chunk][n,i]
__global__ __launch_bounds__(256)
void build_bt(const float* __restrict__ xbc, __bf16* __restrict__ Bt) {
  const int chunk = blockIdx.x;
  for (int e = threadIdx.x; e < DSTATE * QCHUNK; e += 256) {
    int i = e & 63, n = e >> 6;
    Bt[(size_t)chunk * (DSTATE * QCHUNK) + e] =
        tobf(xbc[((size_t)chunk * QCHUNK + i) * CONVDIM + DINNER + n]);
  }
}

// ---------------- K9: per-(chunk,head) transposed X: Xt[p,i], Xdt[p,i]=X*exp(cs63-cs_i)*dt_i
__global__ __launch_bounds__(256)
void build_xt(const float* __restrict__ xbc, const float* __restrict__ dt,
              const float* __restrict__ cs, __bf16* __restrict__ Xt,
              __bf16* __restrict__ Xdt) {
  const int bch = blockIdx.x;
  const int chunk = bch / NHEADS, h = bch % NHEADS;
  const float cs63 = cs[(size_t)bch * QCHUNK + QCHUNK - 1];
  for (int e = threadIdx.x; e < QCHUNK * QCHUNK; e += 256) {
    int i = e & 63, p = e >> 6;
    int row = chunk * QCHUNK + i;
    float xv = xbc[(size_t)row * CONVDIM + h * HEADDIM + p];
    float wt = expf(cs63 - cs[(size_t)bch * QCHUNK + i]) * dt[(size_t)row * NHEADS + h];
    Xt[(size_t)bch * 4096 + e]  = tobf(xv);
    Xdt[(size_t)bch * 4096 + e] = tobf(xv * wt);
  }
}

// ---------------- K10: W[t,i] = G[t,i]*exp(cs_t-cs_i)*dt_i (causal), bf16
__global__ __launch_bounds__(256)
void wbuild(const float* __restrict__ G, const float* __restrict__ cs,
            const float* __restrict__ dt, __bf16* __restrict__ W) {
  const size_t total = (size_t)NBATCH * 4096;
  for (size_t idx = blockIdx.x * (size_t)blockDim.x + threadIdx.x; idx < total;
       idx += (size_t)gridDim.x * blockDim.x) {
    int e = (int)(idx & 4095);
    int bch = (int)(idx >> 12);
    int ti = e >> 6, ii = e & 63;
    int chunk = bch / NHEADS, h = bch % NHEADS;
    float v = 0.f;
    if (ii <= ti)
      v = G[(size_t)chunk * 4096 + e] *
          expf(cs[(size_t)bch * QCHUNK + ti] - cs[(size_t)bch * QCHUNK + ii]) *
          dt[((size_t)chunk * QCHUNK + ii) * NHEADS + h];
    W[idx] = tobf(v);
  }
}

// ---------------- K11: sequential inter-chunk state combine (32 steps)
// Hbf[b,c,h] = state at START of chunk c (bf16); H' = exp(cs63)*H + Spart
__global__ __launch_bounds__(256)
void state_combine(const float* __restrict__ Spart, const float* __restrict__ cs,
                   __bf16* __restrict__ Hbf) {
  const int bh = blockIdx.x;                 // b*NH + h
  const int b = bh / NHEADS, h = bh % NHEADS;
  const int tid = threadIdx.x;
  float H[32];
  #pragma unroll
  for (int j = 0; j < 32; ++j) H[j] = 0.f;
  for (int c = 0; c < NCHUNK; ++c) {
    const int bch = (b * NCHUNK + c) * NHEADS + h;
    const size_t base = (size_t)bch * (HEADDIM * DSTATE);
    #pragma unroll
    for (int j = 0; j < 32; ++j)
      Hbf[base + tid + j * 256] = tobf(H[j]);
    const float dec = expf(cs[(size_t)bch * QCHUNK + QCHUNK - 1]);
    #pragma unroll
    for (int j = 0; j < 32; ++j)
      H[j] = fmaf(dec, H[j], Spart[base + tid + j * 256]);
  }
}

// ---------------- K12: y = (y + D*x) * silu(z); gated rmsnorm; emit bf16
__global__ __launch_bounds__(256)
void gate_norm(const float* __restrict__ y, const float* __restrict__ zx,
               const float* __restrict__ xbc, const float* __restrict__ Dp,
               const float* __restrict__ gw, __bf16* __restrict__ outbf) {
  __shared__ float sg[DINNER];
  __shared__ float red[256];
  const int m = blockIdx.x, tid = threadIdx.x;
  float ss = 0.f;
  for (int i = tid; i < DINNER; i += 256) {
    float z = zx[(size_t)m * DINPROJ + i];
    float yv = y[(size_t)m * DINNER + i] + Dp[i >> 6] * xbc[(size_t)m * CONVDIM + i];
    float g = yv * (z / (1.f + expf(-z)));
    sg[i] = g;
    ss += g * g;
  }
  float tot = block_reduce_sum(ss, red);
  float sc = rsqrtf(tot / (float)DINNER + EPSF);
  for (int i = tid; i < DINNER; i += 256)
    outbf[(size_t)m * DINNER + i] = tobf(sg[i] * sc * gw[i]);
}

// ---------------- host orchestration ----------------
extern "C" void kernel_launch(void* const* d_in, const int* in_sizes, int n_in,
                              void* d_out, int out_size, void* d_ws, size_t ws_size,
                              hipStream_t stream) {
  const float* x       = (const float*)d_in[0];
  const float* norm_w  = (const float*)d_in[1];
  const float* op_w    = (const float*)d_in[2];
  const float* op_b    = (const float*)d_in[3];
  const float* in_w[2]    = {(const float*)d_in[4],  (const float*)d_in[12]};
  const float* conv_w[2]  = {(const float*)d_in[5],  (const float*)d_in[13]};
  const float* conv_b[2]  = {(const float*)d_in[6],  (const float*)d_in[14]};
  const float* dt_bias[2] = {(const float*)d_in[7],  (const float*)d_in[15]};
  const float* A_log[2]   = {(const float*)d_in[8],  (const float*)d_in[16]};
  const float* Dp[2]      = {(const float*)d_in[9],  (const float*)d_in[17]};
  const float* gnorm[2]   = {(const float*)d_in[10], (const float*)d_in[18]};
  const float* outp_w[2]  = {(const float*)d_in[11], (const float*)d_in[19]};
  float* out = (float*)d_out;

  char* p = (char*)d_ws;
  auto alloc = [&](size_t bytes) -> void* {
    void* r = (void*)p;
    p += (bytes + 255) & ~(size_t)255;
    return r;
  };

  __bf16* u_bf[2];
  u_bf[0] = (__bf16*)alloc((size_t)MROWS * DHALF * 2);
  u_bf[1] = (__bf16*)alloc((size_t)MROWS * DHALF * 2);
  __bf16* w_in_bf[2];
  w_in_bf[0] = (__bf16*)alloc((size_t)DINPROJ * DHALF * 2);
  w_in_bf[1] = (__bf16*)alloc((size_t)DINPROJ * DHALF * 2);
  __bf16* w_outp_bf[2];
  w_outp_bf[0] = (__bf16*)alloc((size_t)DHALF * DINNER * 2);
  w_outp_bf[1] = (__bf16*)alloc((size_t)DHALF * DINNER * 2);
  __bf16* w_op_bf = (__bf16*)alloc((size_t)DMODEL * DMODEL * 2);

  float* zx      = (float*)alloc((size_t)MROWS * DINPROJ * 4);
  float* xbc     = (float*)alloc((size_t)MROWS * CONVDIM * 4);
  float* dtb     = (float*)alloc((size_t)MROWS * NHEADS * 4);
  float* ybuf    = (float*)alloc((size_t)MROWS * DINNER * 4);
  __bf16* yn_bf  = (__bf16*)alloc((size_t)MROWS * DINNER * 2);
  float* catbuf  = (float*)alloc((size_t)MROWS * DMODEL * 4);
  __bf16* cat_bf = (__bf16*)alloc((size_t)MROWS * DMODEL * 2);

  // chunked-SSD scratch
  float*  csbuf = (float*)alloc((size_t)NBATCH * QCHUNK * 4);
  __bf16* Bbf   = (__bf16*)alloc((size_t)MROWS * DSTATE * 2);
  __bf16* Cbf   = (__bf16*)alloc((size_t)MROWS * DSTATE * 2);
  __bf16* Btbf  = (__bf16*)alloc((size_t)BB * NCHUNK * DSTATE * QCHUNK * 2);
  __bf16* Xtbf  = (__bf16*)alloc((size_t)NBATCH * 4096 * 2);
  __bf16* Xdtbf = (__bf16*)alloc((size_t)NBATCH * 4096 * 2);
  float*  Gbuf  = (float*)alloc((size_t)BB * NCHUNK * 4096 * 4);
  __bf16* Wbf   = (__bf16*)alloc((size_t)NBATCH * 4096 * 2);
  float*  Spart = (float*)alloc((size_t)NBATCH * HEADDIM * DSTATE * 4);
  __bf16* Hbf   = (__bf16*)alloc((size_t)NBATCH * HEADDIM * DSTATE * 2);

  // 1. rmsnorm + split + backward time-flip
  rmsnorm_split<<<MROWS, 256, 0, stream>>>(x, norm_w, u_bf[0], u_bf[1]);

  // 2. weight conversions
  cvt_bf16<<<512, 256, 0, stream>>>(in_w[0],   w_in_bf[0],   (size_t)DINPROJ * DHALF);
  cvt_bf16<<<512, 256, 0, stream>>>(in_w[1],   w_in_bf[1],   (size_t)DINPROJ * DHALF);
  cvt_bf16<<<512, 256, 0, stream>>>(outp_w[0], w_outp_bf[0], (size_t)DHALF * DINNER);
  cvt_bf16<<<512, 256, 0, stream>>>(outp_w[1], w_outp_bf[1], (size_t)DHALF * DINNER);
  cvt_bf16<<<512, 256, 0, stream>>>(op_w,      w_op_bf,      (size_t)DMODEL * DMODEL);

  for (int d = 0; d < 2; ++d) {
    // in-proj
    gemm_bf16<<<dim3((DINPROJ + 63) / 64, MROWS / 64), 128, 0, stream>>>(
        u_bf[d], DHALF, w_in_bf[d], DHALF, zx, DINPROJ,
        DINPROJ, DHALF, nullptr, nullptr, 0, 0);
    conv_silu<<<MROWS, 256, 0, stream>>>(zx, conv_w[d], conv_b[d], xbc);
    dt_prep<<<256, 256, 0, stream>>>(zx, dt_bias[d], dtb);

    // ---- chunked SSD scan ----
    chunk_cs<<<(NBATCH + 255) / 256, 256, 0, stream>>>(dtb, A_log[d], csbuf);
    build_bc<<<MROWS, 128, 0, stream>>>(xbc, Bbf, Cbf);
    build_bt<<<BB * NCHUNK, 256, 0, stream>>>(xbc, Btbf);
    build_xt<<<NBATCH, 256, 0, stream>>>(xbc, dtb, csbuf, Xtbf, Xdtbf);
    // G = C_chunk @ B_chunk^T per (b,chunk): [64,64], K=128
    gemm_bf16_batched<<<dim3(1, BB * NCHUNK), 128, 0, stream>>>(
        Cbf, DSTATE, (size_t)QCHUNK * DSTATE, 1,
        Bbf, DSTATE, (size_t)QCHUNK * DSTATE, 1,
        Gbuf, QCHUNK, (size_t)4096, 1, 1, 0,
        QCHUNK, DSTATE, nullptr, 0);
    wbuild<<<4096, 256, 0, stream>>>(Gbuf, csbuf, dtb, Wbf);
    // Y_intra = W @ X  per (b,chunk,head): out [t,p] into ybuf
    gemm_bf16_batched<<<dim3(1, NBATCH), 128, 0, stream>>>(
        Wbf, QCHUNK, (size_t)4096, 1,
        Xtbf, QCHUNK, (size_t)4096, 1,
        ybuf, DINNER, (size_t)QCHUNK * DINNER, NHEADS, NHEADS, HEADDIM,
        QCHUNK, QCHUNK, nullptr, 0);
    // Spart[p,n] = Xdt^T-form @ B_chunk: [64,128], K=64
    gemm_bf16_batched<<<dim3(DSTATE / 64, NBATCH), 128, 0, stream>>>(
        Xdtbf, QCHUNK, (size_t)4096, 1,
        Btbf, QCHUNK, (size_t)DSTATE * QCHUNK, NHEADS,
        Spart, DSTATE, (size_t)HEADDIM * DSTATE, 1, 1, 0,
        DSTATE, QCHUNK, nullptr, 0);
    state_combine<<<BB * NHEADS, 256, 0, stream>>>(Spart, csbuf, Hbf);
    // Y_inter = exp(cs_t) * (C_chunk @ H_prev^T): accumulate into ybuf
    gemm_bf16_batched<<<dim3(1, NBATCH), 128, 0, stream>>>(
        Cbf, DSTATE, (size_t)QCHUNK * DSTATE, NHEADS,
        Hbf, DSTATE, (size_t)HEADDIM * DSTATE, 1,
        ybuf, DINNER, (size_t)QCHUNK * DINNER, NHEADS, NHEADS, HEADDIM,
        HEADDIM, DSTATE, csbuf, 1);
    // ---- end scan ----

    gate_norm<<<MROWS, 256, 0, stream>>>(ybuf, zx, xbc, Dp[d], gnorm[d], yn_bf);
    // out-proj into cat[:, d*512:], flip rows for backward direction
    gemm_bf16<<<dim3(DHALF / 64, MROWS / 64), 128, 0, stream>>>(
        yn_bf, DINNER, w_outp_bf[d], DINNER, catbuf, DMODEL,
        DHALF, DINNER, nullptr, nullptr, d * DHALF, d == 1);
  }

  // final projection with bias + residual
  cvt_bf16<<<1024, 256, 0, stream>>>(catbuf, cat_bf, (size_t)MROWS * DMODEL);
  gemm_bf16<<<dim3(DMODEL / 64, MROWS / 64), 128, 0, stream>>>(
      cat_bf, DMODEL, w_op_bf, DMODEL, out, DMODEL,
      DMODEL, DMODEL, op_b, x, 0, 0);
}
